// CSM_32590211842133
// MI455X (gfx1250) — compile-verified, matrix operand order checked by reading, not run on hardware
//
#include <hip/hip_runtime.h>

// Shapes from the reference
#define BB   16
#define CC   128     // channels == Mamba sequence length L
#define NN   2000    // spatial == Mamba feature dim
#define DIM  4000    // DI = 2*N
#define DTR  125     // dt_rank
#define SSZ  8       // state size
#define XDW  141     // DT_RANK + 2*S
#define NNP  2016    // NN padded to multiple of 32
#define DTRP 128     // DTR padded to multiple of 32

typedef __attribute__((ext_vector_type(16))) __bf16 v16bf;
typedef __attribute__((ext_vector_type(8)))  float  v8f;

// ---------------- f32 -> bf16 (round-to-nearest-even), proven to compile ----
__device__ __forceinline__ __bf16 f2bf(float x) {
  union { float f; unsigned u; } v; v.f = x;
  unsigned r = v.u + 0x7FFFu + ((v.u >> 16) & 1u);
  union { unsigned short s; __bf16 b; } o; o.s = (unsigned short)(r >> 16);
  return o.b;
}

// ---------------- WMMA bf16 fragment: two 16B b128 loads per lane -----------
// 16-bit 16x32 A / 32x16 B per-lane layout (CDNA5 ISA §7.12.2):
// lane L -> row (L&15); elements 0..7 = K kb..kb+7, 8..15 = K kb+16..kb+23,
// kb = (L>>4)*8.  p must point at (row, kb), 16B-aligned.
__device__ __forceinline__ v16bf ldfrag(const __bf16* p) {
  union { uint4 q[2]; v16bf v; } u;
  u.q[0] = *(const uint4*)(p);
  u.q[1] = *(const uint4*)(p + 16);
  return u.v;
}

// ---------------- GEMM: O[r,c] = sum_k A[r,k]*W[c,k] (+bias[c]) -------------
// A,W bf16, K-contiguous, Kp multiple of 32 (zero padded).  256 thr = 8 waves,
// wave computes 32x64 via 8 accumulators (2 A-frags x 4 B-frags), block 256x64.
// Out-of-range rows/cols are clamped on load (harmless) and masked on store.
__global__ void __launch_bounds__(256)
gemm_bf16(const __bf16* __restrict__ A, long aRS, long aBS,
          const __bf16* __restrict__ W, long wRS,
          const float* __restrict__ bias,
          float* __restrict__ O, long oRS, long oCS, long oBS,
          int R, int Cn, int Kp) {
  int lane  = threadIdx.x & 31;
  int wave  = threadIdx.x >> 5;
  int r0    = blockIdx.y * 256 + wave * 32;
  int c0    = blockIdx.x * 64;
  int rl    = lane & 15;
  int klane = (lane >> 4) << 3;

  const __bf16* Ab  = A + (long)blockIdx.z * aBS;
  const __bf16* pa0 = Ab + (long)min(r0 + rl,      R - 1) * aRS + klane;
  const __bf16* pa1 = Ab + (long)min(r0 + 16 + rl, R - 1) * aRS + klane;
  const __bf16* pw0 = W  + (long)min(c0 + rl,      Cn - 1) * wRS + klane;
  const __bf16* pw1 = W  + (long)min(c0 + 16 + rl, Cn - 1) * wRS + klane;
  const __bf16* pw2 = W  + (long)min(c0 + 32 + rl, Cn - 1) * wRS + klane;
  const __bf16* pw3 = W  + (long)min(c0 + 48 + rl, Cn - 1) * wRS + klane;

  v8f acc[2][4] = {};
  for (int k0 = 0; k0 < Kp; k0 += 32) {
    v16bf a0 = ldfrag(pa0 + k0);
    v16bf a1 = ldfrag(pa1 + k0);
    v16bf w0 = ldfrag(pw0 + k0);
    v16bf w1 = ldfrag(pw1 + k0);
    v16bf w2 = ldfrag(pw2 + k0);
    v16bf w3 = ldfrag(pw3 + k0);
    acc[0][0] = __builtin_amdgcn_wmma_f32_16x16x32_bf16(false, a0, false, w0, (short)0, acc[0][0], false, false);
    acc[0][1] = __builtin_amdgcn_wmma_f32_16x16x32_bf16(false, a0, false, w1, (short)0, acc[0][1], false, false);
    acc[0][2] = __builtin_amdgcn_wmma_f32_16x16x32_bf16(false, a0, false, w2, (short)0, acc[0][2], false, false);
    acc[0][3] = __builtin_amdgcn_wmma_f32_16x16x32_bf16(false, a0, false, w3, (short)0, acc[0][3], false, false);
    acc[1][0] = __builtin_amdgcn_wmma_f32_16x16x32_bf16(false, a1, false, w0, (short)0, acc[1][0], false, false);
    acc[1][1] = __builtin_amdgcn_wmma_f32_16x16x32_bf16(false, a1, false, w1, (short)0, acc[1][1], false, false);
    acc[1][2] = __builtin_amdgcn_wmma_f32_16x16x32_bf16(false, a1, false, w2, (short)0, acc[1][2], false, false);
    acc[1][3] = __builtin_amdgcn_wmma_f32_16x16x32_bf16(false, a1, false, w3, (short)0, acc[1][3], false, false);
  }

  // D layout: element e -> row tileBase + e + (lane>>4)*8, col c0+j*16+(lane&15)
  float* Ob = O + (long)blockIdx.z * oBS;
#pragma unroll
  for (int t = 0; t < 2; ++t) {
    int rbase = r0 + t * 16 + ((lane >> 4) << 3);
#pragma unroll
    for (int j = 0; j < 4; ++j) {
      int cc = c0 + j * 16 + rl;
      if (cc < Cn) {
        float bd = bias ? bias[cc] : 0.f;
        long cadr = (long)cc * oCS;
#pragma unroll
        for (int e = 0; e < 8; ++e) {
          int rr = rbase + e;
          if (rr < R) Ob[(long)rr * oRS + cadr] = acc[t][j][e] + bd;
        }
      }
    }
  }
}

// ---------------- per-row mean / variance (rows of length n) ----------------
__global__ void row_stats(const float* __restrict__ in, float* __restrict__ mean,
                          float* __restrict__ var, int n) {
  __shared__ float sh[256], sh2[256];
  long row = blockIdx.x;
  const float* p = in + row * n;
  float s = 0.f, s2 = 0.f;
  for (int i = threadIdx.x; i < n; i += 256) { float v = p[i]; s += v; s2 += v * v; }
  sh[threadIdx.x] = s; sh2[threadIdx.x] = s2;
  __syncthreads();
  for (int o = 128; o > 0; o >>= 1) {
    if ((int)threadIdx.x < o) { sh[threadIdx.x] += sh[threadIdx.x + o]; sh2[threadIdx.x] += sh2[threadIdx.x + o]; }
    __syncthreads();
  }
  if (threadIdx.x == 0) {
    float m = sh[0] / n;
    mean[row] = m;
    var[row]  = sh2[0] / n - m * m;
  }
}

// inorm output has exactly zero mean per (b,c) => batch mean is 0, batch var per
// channel is mean_b[ v/(v+1e-3) ].  Emit fused per-(b,c) scale/shift.
__global__ void bn_coef(const float* __restrict__ m, const float* __restrict__ v,
                        const float* __restrict__ g, const float* __restrict__ beta,
                        float* __restrict__ s, float* __restrict__ t) {
  int c = threadIdx.x;           // 0..127
  float acc = 0.f;
  for (int b = 0; b < BB; ++b) {
    float vb = v[b * CC + c];
    acc += vb / (vb + 1e-3f);
  }
  float bn = rsqrtf(acc / BB + 1e-5f) * g[c];
  for (int b = 0; b < BB; ++b) {
    int i = b * CC + c;
    float sc = rsqrtf(v[i] + 1e-3f) * bn;
    s[i] = sc;
    t[i] = beta[c] - m[i] * sc;
  }
}

// fused norm+ReLU, transposed to (b, n, c) bf16 so the CxC conv GEMM gets
// K-contiguous operands.
__global__ void norm_relu_t(const float* __restrict__ in, const float* __restrict__ s,
                            const float* __restrict__ t, __bf16* __restrict__ o,
                            long total) {            // total = BB*NN*CC
  long j = (long)blockIdx.x * 256 + threadIdx.x;
  if (j >= total) return;
  int  c  = (int)(j % CC);
  long bn = j / CC;
  int  n  = (int)(bn % NN);
  long b  = bn / NN;
  long row = b * CC + c;
  float v = in[b * (long)CC * NN + (long)c * NN + n] * s[row] + t[row];
  o[j] = f2bf(v > 0.f ? v : 0.f);
}

// LayerNorm over N, writing K-padded bf16 rows (2048 x NNP) directly.
__global__ void ln_apply_bf(const float* __restrict__ in, const float* __restrict__ mean,
                            const float* __restrict__ var, const float* __restrict__ w,
                            const float* __restrict__ bsh, __bf16* __restrict__ o,
                            long total) {            // total = 2048*NNP
  long i = (long)blockIdx.x * 256 + threadIdx.x;
  if (i >= total) return;
  long r = i / NNP;
  int  k = (int)(i - r * NNP);
  float v = 0.f;
  if (k < NN) v = (in[r * NN + k] - mean[r]) * rsqrtf(var[r] + 1e-5f) * w[k] + bsh[k];
  o[i] = f2bf(v);
}

// generic f32 -> bf16 convert with row restride + zero K-padding
__global__ void cvt_pad(const float* __restrict__ src, long srcRS, int srcK,
                        __bf16* __restrict__ dst, int dstK, long total) {
  long i = (long)blockIdx.x * 256 + threadIdx.x;
  if (i >= total) return;
  long r = i / dstK;
  int  k = (int)(i - r * dstK);
  float v = (k < srcK) ? src[r * srcRS + k] : 0.f;
  dst[i] = f2bf(v);
}

// depthwise causal conv (DCONV=4) along L + SiLU; dual-store f32 (scan) + bf16 (GEMM)
__global__ void dwconv_silu(const float* __restrict__ xz, const float* __restrict__ cw,
                            const float* __restrict__ cb, float* __restrict__ xc,
                            __bf16* __restrict__ xcbf, long total) {
  long i = (long)blockIdx.x * 256 + threadIdx.x;
  if (i >= total) return;
  int  d  = (int)(i % DIM);
  long bl = i / DIM;
  int  l  = (int)(bl % CC);
  long b  = bl / CC;
  const float* base = xz + (b * CC) * (long)(2 * DIM) + d;   // xi = xz[..., :DIM]
  float acc = cb[d];
#pragma unroll
  for (int k = 0; k < 4; ++k) {
    int ll = l - 3 + k;
    if (ll >= 0) acc += base[(long)ll * (2 * DIM)] * cw[d * 4 + k];
  }
  float v = acc / (1.f + __expf(-acc));    // SiLU
  xc[i]   = v;
  xcbf[i] = f2bf(v);
}

// selective scan: 64000 independent (b,d) lanes, 128 sequential steps.
// softplus(dt), S=8 recurrence, D_p skip, silu(z) gate; writes bf16 y directly.
__global__ void scan_kernel(const float* __restrict__ dtpre, const float* __restrict__ xc,
                            const float* __restrict__ xdbl, const float* __restrict__ xz,
                            const float* __restrict__ A_log, const float* __restrict__ D_p,
                            __bf16* __restrict__ ybf) {
  int i = blockIdx.x * 256 + threadIdx.x;    // < BB*DIM = 64000
  if (i >= BB * DIM) return;
  int d = i % DIM;
  int b = i / DIM;
  float A[SSZ];
#pragma unroll
  for (int s = 0; s < SSZ; ++s) A[s] = -__expf(A_log[d * SSZ + s]);
  float Dp = D_p[d];
  float h[SSZ] = {};
  long rbase = (long)b * CC;
  for (int l = 0; l < CC; ++l) {
    long row = rbase + l;
    float pre = dtpre[row * DIM + d];
    float dt  = pre > 20.f ? pre : log1pf(__expf(pre));   // softplus
    float u   = xc[row * DIM + d];
    const float* xd = xdbl + row * XDW;
    float du = dt * u;
    float y = 0.f;
#pragma unroll
    for (int s = 0; s < SSZ; ++s) {
      float hh = __expf(dt * A[s]) * h[s] + du * xd[DTR + s];   // B_l
      h[s] = hh;
      y += hh * xd[DTR + SSZ + s];                               // C_l
    }
    float z  = xz[row * (long)(2 * DIM) + DIM + d];
    float sz = z / (1.f + __expf(-z));
    ybf[row * DIM + d] = f2bf((y + u * Dp) * sz);
  }
}

__global__ void add_kernel(const float* __restrict__ a, const float* __restrict__ b,
                           float* __restrict__ o, long n) {
  long i = (long)blockIdx.x * 256 + threadIdx.x;
  if (i < n) o[i] = a[i] + b[i];
}

// ---------------------------------------------------------------------------
extern "C" void kernel_launch(void* const* d_in, const int* in_sizes, int n_in,
                              void* d_out, int out_size, void* d_ws, size_t ws_size,
                              hipStream_t stream) {
  (void)in_sizes; (void)n_in; (void)out_size; (void)ws_size;
  const float* x       = (const float*)d_in[0];
  const float* bn1_g   = (const float*)d_in[1];
  const float* bn1_b   = (const float*)d_in[2];
  const float* conv1_w = (const float*)d_in[3];
  const float* conv1_b = (const float*)d_in[4];
  const float* ln_w    = (const float*)d_in[5];
  const float* ln_b    = (const float*)d_in[6];
  const float* W_in    = (const float*)d_in[7];
  const float* convm_w = (const float*)d_in[8];
  const float* convm_b = (const float*)d_in[9];
  const float* W_x     = (const float*)d_in[10];
  const float* W_dt    = (const float*)d_in[11];
  const float* b_dt    = (const float*)d_in[12];
  const float* A_log   = (const float*)d_in[13];
  const float* D_p     = (const float*)d_in[14];
  const float* W_out   = (const float*)d_in[15];
  const float* bn3_g   = (const float*)d_in[16];
  const float* bn3_b   = (const float*)d_in[17];
  const float* conv3_w = (const float*)d_in[18];
  const float* conv3_b = (const float*)d_in[19];
  float* out = (float*)d_out;

  const long TOT  = (long)BB * CC * NN;            // 4,096,000
  const long ROWS = (long)BB * CC;                 // 2048

  // ---- workspace carve-out (256B aligned slices) ----
  char* base = (char*)d_ws;
  size_t off = 0;
  auto allocF = [&](long n) { float*  p = (float*) (base + off); off = (off + n * 4  + 255) & ~(size_t)255; return p; };
  auto allocB = [&](long n) { __bf16* p = (__bf16*)(base + off); off = (off + n * 2  + 255) & ~(size_t)255; return p; };

  float* stat_m = allocF(ROWS);
  float* stat_v = allocF(ROWS);
  float* coef_s = allocF(ROWS);
  float* coef_t = allocF(ROWS);
  float* out1   = allocF(TOT);                 // block1 output (kept for residual)
  float* xz     = allocF(ROWS * 2 * DIM);      // 2048 x 8000; front 8.2M reused later
  float* xc     = allocF(ROWS * DIM);          // xc f32 (scan input u)
  float* xdbl   = allocF(ROWS * XDW);          // 2048 x 141
  float* dtp    = allocF(ROWS * DIM);          // dt_pre; later conv3 raw output
  __bf16* ht    = allocB(TOT);                 // (b,n,c) bf16 act for conv GEMMs; reused for h3
  __bf16* lnbf  = allocB(ROWS * NNP);          // LN out, padded K
  __bf16* xcbf  = allocB(ROWS * DIM);          // xc bf16
  __bf16* xdbf  = allocB(ROWS * DTRP);         // xdbl[:, :125] bf16, padded K
  __bf16* ybf   = allocB(ROWS * DIM);          // scan output bf16
  __bf16* wbf_in  = allocB((long)2 * DIM * NNP);   // 8000 x 2016
  __bf16* wbf_out = allocB((long)NN * DIM);        // 2000 x 4000
  __bf16* wbf_x   = allocB((long)XDW * DIM);       // 141 x 4000
  __bf16* wbf_dt  = allocB((long)DIM * DTRP);      // 4000 x 128
  __bf16* wbf_c1  = allocB(CC * CC);
  __bf16* wbf_c3  = allocB(CC * CC);
  // reuse: after the scan, xz is dead -> out_m and pre3 live in its footprint
  float* outm = xz;                            // 4.096M
  float* pre3 = xz + TOT;                      // 4.096M

  auto eb = [](long n) { return (unsigned)((n + 255) / 256); };

  // ---- one-time weight conversions to padded bf16 ----
  cvt_pad<<<eb((long)2 * DIM * NNP), 256, 0, stream>>>(W_in,  NN,  NN,  wbf_in,  NNP, (long)2 * DIM * NNP);
  cvt_pad<<<eb((long)NN * DIM),      256, 0, stream>>>(W_out, DIM, DIM, wbf_out, DIM, (long)NN * DIM);
  cvt_pad<<<eb((long)XDW * DIM),     256, 0, stream>>>(W_x,   DIM, DIM, wbf_x,   DIM, (long)XDW * DIM);
  cvt_pad<<<eb((long)DIM * DTRP),    256, 0, stream>>>(W_dt,  DTR, DTR, wbf_dt,  DTRP, (long)DIM * DTRP);
  cvt_pad<<<eb(CC * CC),             256, 0, stream>>>(conv1_w, CC, CC, wbf_c1,  CC, CC * CC);
  cvt_pad<<<eb(CC * CC),             256, 0, stream>>>(conv3_w, CC, CC, wbf_c3,  CC, CC * CC);

  // ---- block 1: inorm+bnorm+relu (fused coefs), CxC conv via WMMA ----
  row_stats<<<(unsigned)ROWS, 256, 0, stream>>>(x, stat_m, stat_v, NN);
  bn_coef<<<1, CC, 0, stream>>>(stat_m, stat_v, bn1_g, bn1_b, coef_s, coef_t);
  norm_relu_t<<<eb(TOT), 256, 0, stream>>>(x, coef_s, coef_t, ht, TOT);
  {
    dim3 g(2, 8, BB);   // Cn=128, R=2000 (per batch), 256 rows/block
    gemm_bf16<<<g, 256, 0, stream>>>(ht, CC, (long)NN * CC,
                                     wbf_c1, CC, conv1_b,
                                     out1, 1, NN, (long)CC * NN,
                                     NN, CC, CC);
  }

  // ---- LayerNorm over N ----
  row_stats<<<(unsigned)ROWS, 256, 0, stream>>>(out1, stat_m, stat_v, NN);
  ln_apply_bf<<<eb(ROWS * NNP), 256, 0, stream>>>(out1, stat_m, stat_v, ln_w, ln_b,
                                                  lnbf, ROWS * NNP);

  // ---- xz = u @ W_in^T   (2048 x 8000 x K=2016, ~65 GFLOP) ----
  {
    dim3 g(125, 8, 1);
    gemm_bf16<<<g, 256, 0, stream>>>(lnbf, NNP, 0, wbf_in, NNP, nullptr,
                                     xz, 2 * DIM, 1, 0,
                                     (int)ROWS, 2 * DIM, NNP);
  }
  // depthwise conv + SiLU (f32 + bf16 outputs)
  dwconv_silu<<<eb(ROWS * DIM), 256, 0, stream>>>(xz, convm_w, convm_b, xc, xcbf, ROWS * DIM);
  // xdbl = xc @ W_x^T  (Cn=141)
  {
    dim3 g(3, 8, 1);
    gemm_bf16<<<g, 256, 0, stream>>>(xcbf, DIM, 0, wbf_x, DIM, nullptr,
                                     xdbl, XDW, 1, 0,
                                     (int)ROWS, XDW, DIM);
  }
  // bf16 view of xdbl[:, :125], padded to 128
  cvt_pad<<<eb(ROWS * DTRP), 256, 0, stream>>>(xdbl, XDW, DTR, xdbf, DTRP, ROWS * DTRP);
  // dt_pre = xdbl[:, :125] @ W_dt^T + b_dt  (Cn=4000, K=128)
  {
    dim3 g(63, 8, 1);
    gemm_bf16<<<g, 256, 0, stream>>>(xdbf, DTRP, 0, wbf_dt, DTRP, b_dt,
                                     dtp, DIM, 1, 0,
                                     (int)ROWS, DIM, DTRP);
  }
  // selective scan -> bf16 y
  scan_kernel<<<eb((long)BB * DIM), 256, 0, stream>>>(dtp, xc, xdbl, xz, A_log, D_p, ybf);
  // out_m = y @ W_out^T  (2048 x 2000 x K=4000, ~33 GFLOP); xz now dead
  {
    dim3 g(32, 8, 1);
    gemm_bf16<<<g, 256, 0, stream>>>(ybf, DIM, 0, wbf_out, DIM, nullptr,
                                     outm, NN, 1, 0,
                                     (int)ROWS, NN, DIM);
  }
  // pre3 = out1 + out_m
  add_kernel<<<eb(TOT), 256, 0, stream>>>(out1, outm, pre3, TOT);

  // ---- block 3 + final residual ----
  row_stats<<<(unsigned)ROWS, 256, 0, stream>>>(pre3, stat_m, stat_v, NN);
  bn_coef<<<1, CC, 0, stream>>>(stat_m, stat_v, bn3_g, bn3_b, coef_s, coef_t);
  norm_relu_t<<<eb(TOT), 256, 0, stream>>>(pre3, coef_s, coef_t, ht, TOT);
  {
    dim3 g(2, 8, BB);
    gemm_bf16<<<g, 256, 0, stream>>>(ht, CC, (long)NN * CC,
                                     wbf_c3, CC, conv3_b,
                                     dtp, 1, NN, (long)CC * NN,   // conv3 raw -> dtp slot
                                     NN, CC, CC);
  }
  add_kernel<<<eb(TOT), 256, 0, stream>>>(dtp, x, out, TOT);
}
